// AdvancedGCNRegression_91328184582218
// MI455X (gfx1250) — matrix operand order repaired and checked by reading, not compiled, not run on hardware
//
#include <hip/hip_runtime.h>
#include <hip/hip_bf16.h>

#define N_NODES 100000
#define N_EDGES 1600000
#define ET      (N_EDGES + N_NODES)   // edges + self loops
#define NEG_SLOPE 0.2f

typedef __attribute__((ext_vector_type(16))) _Float16 v16h;
typedef __attribute__((ext_vector_type(8)))  float    v8f;

// ---------------------------------------------------------------- utilities
__global__ __launch_bounds__(256) void fill_kernel(float* p, float v, int n) {
  int i = blockIdx.x * blockDim.x + threadIdx.x;
  if (i < n) p[i] = v;
}

__global__ __launch_bounds__(256) void deg_kernel(const int* __restrict__ ei,
                                                  float* __restrict__ deg) {
  int e = blockIdx.x * blockDim.x + threadIdx.x;
  if (e < N_EDGES) atomicAdd(&deg[ei[N_EDGES + e]], 1.0f);
}

__global__ __launch_bounds__(256) void rsqrt_kernel(float* d) {
  int i = blockIdx.x * blockDim.x + threadIdx.x;
  if (i < N_NODES) d[i] = rsqrtf(d[i]);
}

__device__ inline void edge_sd(const int* __restrict__ ei, int e, int& s, int& d) {
  if (e < N_EDGES) { s = ei[e]; d = ei[N_EDGES + e]; }
  else             { s = e - N_EDGES; d = s; }
}

__device__ inline void atomicMaxF(float* addr, float val) {
  int old = *(volatile int*)addr;
  while (__int_as_float(old) < val) {
    int assumed = old;
    old = atomicCAS((int*)addr, assumed, __float_as_int(val));
    if (old == assumed) break;
  }
}

// ------------------------------------------------------------ WMMA GEMM
// D[M,C] = A[M,K] @ B[K,C]  (f32 in/out, f16 multiply, f32 accumulate)
// - B is staged once per block into LDS *pre-swizzled into WMMA fragment
//   layout* (frag-major, lane-major, 16 contiguous halves per lane) so a
//   fragment load is 2x ds_load_b128.
// - Each wave hoists all B fragments into registers, then computes R
//   consecutive 16-row strips (C/16 accumulators each), reusing register B.
template <int K, int C, int R>
__global__ __launch_bounds__(128) void wmma_gemm_t(
    const float* __restrict__ A, const float* __restrict__ Bw,
    const float* __restrict__ bias, float* __restrict__ D,
    int ntm, int fuse)
{
  constexpr int KT = K / 32;          // k-tiles
  constexpr int TC = C / 16;          // column tiles
  constexpr int NF = KT * TC;         // B fragments
  __shared__ v16h Bs[NF * 32];

  // cooperative stage: global f32 B -> LDS f16 in fragment layout.
  // flat j = ((frag*32 + lane)*16 + e); frag = kt*TC + tn
  // lane: half=lane>>4 selects K group, idx=lane&15 selects column
  _Float16* bsh = (_Float16*)Bs;
  for (int j = threadIdx.x; j < K * C; j += 128) {
    int e  = j & 15;
    int l  = (j >> 4) & 31;
    int f  = j >> 9;
    int kt = f / TC, tn = f % TC;
    int row = kt * 32 + (e >> 3) * 16 + (l >> 4) * 8 + (e & 7);
    int col = tn * 16 + (l & 15);
    bsh[j] = (_Float16)Bw[row * C + col];
  }
  __syncthreads();

  int wave = threadIdx.x >> 5;
  int lane = threadIdx.x & 31;
  int half = lane >> 4, idx = lane & 15;

  // hoist B fragments to registers (2x ds_load_b128 each)
  v16h bf[NF];
#pragma unroll
  for (int f = 0; f < NF; ++f) bf[f] = Bs[f * 32 + lane];

  int tmb = (blockIdx.x * 4 + wave) * R;
#pragma unroll
  for (int r = 0; r < R; ++r) {
    int tm = tmb + r;
    if (tm >= ntm) return;                     // wave-uniform guard
    const float* arow = A + (size_t)(tm * 16 + idx) * K + half * 8;

    v8f acc[TC];
#pragma unroll
    for (int tn = 0; tn < TC; ++tn) acc[tn] = (v8f){};

#pragma unroll
    for (int kt = 0; kt < KT; ++kt) {
      // A fragment: 16x32 f16 (half-wave selects K groups)
      v16h a;
      int k0 = kt * 32;
#pragma unroll
      for (int e = 0; e < 8; ++e) {
        a[e]     = (_Float16)arow[k0 + e];
        a[8 + e] = (_Float16)arow[k0 + 16 + e];
      }
#pragma unroll
      for (int tn = 0; tn < TC; ++tn)
        acc[tn] = __builtin_amdgcn_wmma_f32_16x16x32_f16(
            false, a, false, bf[kt * TC + tn], (short)0, acc[tn], false, false);
    }

#pragma unroll
    for (int tn = 0; tn < TC; ++tn) {
      int col = tn * 16 + idx;
      float bv = fuse ? bias[col] : 0.0f;
#pragma unroll
      for (int q = 0; q < 8; ++q) {
        int row = tm * 16 + q + 8 * half;      // C/D layout: VGPR q <-> row q+8*half
        float v = acc[tn][q] + bv;
        if (fuse) v = fmaxf(v, 0.0f);
        D[(size_t)row * C + col] = v;
      }
    }
  }
}

// ------------------------------------------------------------ GCN aggregate
// Acc[d] += P[s] * dinv[s]*dinv[d]  over all edges incl. self loops.
// Thread = (edge, 4-channel chunk); adjacent threads -> coalesced 16B gathers.
__global__ __launch_bounds__(256) void gcn_agg_kernel(
    const int* __restrict__ ei, const float* __restrict__ P,
    const float* __restrict__ dinv, float* __restrict__ Acc,
    int logChunks, int C)
{
  int t = blockIdx.x * blockDim.x + threadIdx.x;
  int e = t >> logChunks;
  if (e >= ET) return;
  int chunk = t & ((1 << logChunks) - 1);
  int s, d; edge_sd(ei, e, s, d);
  float norm = dinv[s] * dinv[d];
  const float4 p = *(const float4*)(P + (size_t)s * C + chunk * 4);
  float* dst = Acc + (size_t)d * C + chunk * 4;
  atomicAdd(dst + 0, p.x * norm);
  atomicAdd(dst + 1, p.y * norm);
  atomicAdd(dst + 2, p.z * norm);
  atomicAdd(dst + 3, p.w * norm);
}

__global__ __launch_bounds__(256) void bias_relu_kernel(
    float* __restrict__ A, const float* __restrict__ bias, int Cmask, int n)
{
  int i = blockIdx.x * blockDim.x + threadIdx.x;
  if (i < n) A[i] = fmaxf(A[i] + bias[i & Cmask], 0.0f);
}

__global__ __launch_bounds__(256) void add_kernel(
    float* __restrict__ out, const float* __restrict__ a, int n)
{
  int i = blockIdx.x * blockDim.x + threadIdx.x;
  if (i < n) out[i] += a[i];
}

// ------------------------------------------------------------ GAT pieces
__global__ __launch_bounds__(256) void gat_att_kernel(
    const float* __restrict__ hg, const float* __restrict__ att_src,
    const float* __restrict__ att_dst, float* __restrict__ a_s,
    float* __restrict__ a_d)
{
  int t = blockIdx.x * blockDim.x + threadIdx.x;   // node*4 + head
  if (t >= N_NODES * 4) return;
  int h = t & 3, node = t >> 2;
  const float* hp = hg + (size_t)node * 64 + h * 16;
  const float* as = att_src + h * 16;
  const float* ad = att_dst + h * 16;
  float ss = 0.f, sd = 0.f;
#pragma unroll
  for (int c = 0; c < 16; ++c) { float v = hp[c]; ss += v * as[c]; sd += v * ad[c]; }
  a_s[t] = ss; a_d[t] = sd;
}

__device__ inline float gat_logit(const float* a_s, const float* a_d,
                                  int s, int d, int h) {
  float v = a_s[s * 4 + h] + a_d[d * 4 + h];
  return v > 0.f ? v : NEG_SLOPE * v;          // leaky_relu
}

__global__ __launch_bounds__(256) void gat_max_kernel(
    const int* __restrict__ ei, const float* __restrict__ a_s,
    const float* __restrict__ a_d, float* __restrict__ emax)
{
  int t = blockIdx.x * blockDim.x + threadIdx.x;   // edge*4 + head
  if (t >= ET * 4) return;
  int h = t & 3, e = t >> 2;
  int s, d; edge_sd(ei, e, s, d);
  atomicMaxF(&emax[d * 4 + h], gat_logit(a_s, a_d, s, d, h));
}

__global__ __launch_bounds__(256) void gat_sum_kernel(
    const int* __restrict__ ei, const float* __restrict__ a_s,
    const float* __restrict__ a_d, const float* __restrict__ emax,
    float* __restrict__ den)
{
  int t = blockIdx.x * blockDim.x + threadIdx.x;   // edge*4 + head
  if (t >= ET * 4) return;
  int h = t & 3, e = t >> 2;
  int s, d; edge_sd(ei, e, s, d);
  float v = gat_logit(a_s, a_d, s, d, h);
  atomicAdd(&den[d * 4 + h], __expf(v - emax[d * 4 + h]));
}

__global__ __launch_bounds__(256) void gat_agg_kernel(
    const int* __restrict__ ei, const float* __restrict__ hg,
    const float* __restrict__ a_s, const float* __restrict__ a_d,
    const float* __restrict__ emax, const float* __restrict__ den,
    float* __restrict__ Acc)
{
  int t = blockIdx.x * blockDim.x + threadIdx.x;   // edge*16 + chunk
  if (t >= ET * 16) return;
  int chunk = t & 15, e = t >> 4;
  int h = chunk >> 2;                              // head = chunk/4 (16 ch/head)
  int s, d; edge_sd(ei, e, s, d);
  float v = gat_logit(a_s, a_d, s, d, h);
  float alpha = __expf(v - emax[d * 4 + h]) / (den[d * 4 + h] + 1e-16f);
  const float4 p = *(const float4*)(hg + (size_t)s * 64 + chunk * 4);
  float* dst = Acc + (size_t)d * 64 + chunk * 4;
  atomicAdd(dst + 0, p.x * alpha);
  atomicAdd(dst + 1, p.y * alpha);
  atomicAdd(dst + 2, p.z * alpha);
  atomicAdd(dst + 3, p.w * alpha);
}

// ------------------------------------------------------------ final linear
__global__ __launch_bounds__(256) void final_kernel(
    const float* __restrict__ h, const float* __restrict__ Wl,
    const float* __restrict__ bl, float* __restrict__ out)
{
  int n = blockIdx.x * blockDim.x + threadIdx.x;
  if (n >= N_NODES) return;
  const float* hp = h + (size_t)n * 64;
  float s = 0.f;
#pragma unroll
  for (int c = 0; c < 64; ++c) s += hp[c] * Wl[c];
  out[n] = fmaxf(s + bl[0], 0.0f);
}

// ================================================================ launcher
extern "C" void kernel_launch(void* const* d_in, const int* in_sizes, int n_in,
                              void* d_out, int out_size, void* d_ws, size_t ws_size,
                              hipStream_t stream)
{
  const float* x    = (const float*)d_in[0];
  const int*   ei   = (const int*)  d_in[1];
  const float* W1   = (const float*)d_in[2];
  const float* b1   = (const float*)d_in[3];
  const float* W2   = (const float*)d_in[4];
  const float* b2   = (const float*)d_in[5];
  const float* W3   = (const float*)d_in[6];
  const float* b3   = (const float*)d_in[7];
  const float* Wg   = (const float*)d_in[8];
  const float* atts = (const float*)d_in[9];
  const float* attd = (const float*)d_in[10];
  const float* bg   = (const float*)d_in[11];
  const float* Wl   = (const float*)d_in[12];
  const float* bl   = (const float*)d_in[13];
  float* out = (float*)d_out;

  float* ws = (float*)d_ws;
  const size_t NB = (size_t)N_NODES * 64;
  float* B0   = ws;                               // 3 rotating [N,64] buffers
  float* B1   = ws + NB;
  float* B2   = ws + 2 * NB;
  float* a_s  = ws + 3 * NB;                      // [N,4]
  float* a_d  = a_s + (size_t)N_NODES * 4;
  float* emax = a_d + (size_t)N_NODES * 4;
  float* den  = emax + (size_t)N_NODES * 4;
  float* dinv = den + (size_t)N_NODES * 4;        // [N]

  auto cdiv = [](long long a, long long b) { return (int)((a + b - 1) / b); };
  const int NTM   = N_NODES / 16;                 // 6250 row tiles (exact)
  const int GEMMB = cdiv(NTM, 16);                // 4 waves x 4 row tiles per block

  // --- degrees (self loop counted via init=1), dinv = rsqrt(deg)
  fill_kernel<<<cdiv(N_NODES, 256), 256, 0, stream>>>(dinv, 1.0f, N_NODES);
  deg_kernel<<<cdiv(N_EDGES, 256), 256, 0, stream>>>(ei, dinv);
  rsqrt_kernel<<<cdiv(N_NODES, 256), 256, 0, stream>>>(dinv);

  // --- GCN layer 1: h1 = relu(agg(x@W1) + b1)   [N,32] -> B1
  wmma_gemm_t<128, 32, 4><<<GEMMB, 128, 0, stream>>>(x, W1, nullptr, B0, NTM, 0);
  fill_kernel<<<cdiv((long long)N_NODES * 32, 256), 256, 0, stream>>>(B1, 0.0f, N_NODES * 32);
  gcn_agg_kernel<<<cdiv((long long)ET * 8, 256), 256, 0, stream>>>(ei, B0, dinv, B1, 3, 32);
  bias_relu_kernel<<<cdiv((long long)N_NODES * 32, 256), 256, 0, stream>>>(B1, b1, 31, N_NODES * 32);

  // --- GCN layer 2: h2 = relu(agg(h1@W2) + b2)  [N,64] -> B0
  wmma_gemm_t<32, 64, 4><<<GEMMB, 128, 0, stream>>>(B1, W2, nullptr, B2, NTM, 0);
  fill_kernel<<<cdiv(NB, 256), 256, 0, stream>>>(B0, 0.0f, (int)NB);
  gcn_agg_kernel<<<cdiv((long long)ET * 16, 256), 256, 0, stream>>>(ei, B2, dinv, B0, 4, 64);
  bias_relu_kernel<<<cdiv(NB, 256), 256, 0, stream>>>(B0, b2, 63, (int)NB);

  // --- GAT 1: h3 = relu(gat(h2)) ; hg -> B1, out -> B2
  wmma_gemm_t<64, 64, 4><<<GEMMB, 128, 0, stream>>>(B0, Wg, nullptr, B1, NTM, 0);
  gat_att_kernel<<<cdiv((long long)N_NODES * 4, 256), 256, 0, stream>>>(B1, atts, attd, a_s, a_d);
  fill_kernel<<<cdiv((long long)N_NODES * 4, 256), 256, 0, stream>>>(emax, -3.0e38f, N_NODES * 4);
  fill_kernel<<<cdiv((long long)N_NODES * 4, 256), 256, 0, stream>>>(den, 0.0f, N_NODES * 4);
  gat_max_kernel<<<cdiv((long long)ET * 4, 256), 256, 0, stream>>>(ei, a_s, a_d, emax);
  gat_sum_kernel<<<cdiv((long long)ET * 4, 256), 256, 0, stream>>>(ei, a_s, a_d, emax, den);
  fill_kernel<<<cdiv(NB, 256), 256, 0, stream>>>(B2, 0.0f, (int)NB);
  gat_agg_kernel<<<cdiv((long long)ET * 16, 256), 256, 0, stream>>>(ei, B1, a_s, a_d, emax, den, B2);
  bias_relu_kernel<<<cdiv(NB, 256), 256, 0, stream>>>(B2, bg, 63, (int)NB);

  // --- GCN layer 3 + residual: h4 = h3 + relu(agg(h3@W3)+b3) ; -> B2
  wmma_gemm_t<64, 64, 4><<<GEMMB, 128, 0, stream>>>(B2, W3, nullptr, B0, NTM, 0);
  fill_kernel<<<cdiv(NB, 256), 256, 0, stream>>>(B1, 0.0f, (int)NB);
  gcn_agg_kernel<<<cdiv((long long)ET * 16, 256), 256, 0, stream>>>(ei, B0, dinv, B1, 4, 64);
  bias_relu_kernel<<<cdiv(NB, 256), 256, 0, stream>>>(B1, b3, 63, (int)NB);
  add_kernel<<<cdiv(NB, 256), 256, 0, stream>>>(B2, B1, (int)NB);

  // --- GAT 2 (same weights): h5 = relu(gat(h4)) ; hg -> B0, out -> B1
  wmma_gemm_t<64, 64, 4><<<GEMMB, 128, 0, stream>>>(B2, Wg, nullptr, B0, NTM, 0);
  gat_att_kernel<<<cdiv((long long)N_NODES * 4, 256), 256, 0, stream>>>(B0, atts, attd, a_s, a_d);
  fill_kernel<<<cdiv((long long)N_NODES * 4, 256), 256, 0, stream>>>(emax, -3.0e38f, N_NODES * 4);
  fill_kernel<<<cdiv((long long)N_NODES * 4, 256), 256, 0, stream>>>(den, 0.0f, N_NODES * 4);
  gat_max_kernel<<<cdiv((long long)ET * 4, 256), 256, 0, stream>>>(ei, a_s, a_d, emax);
  gat_sum_kernel<<<cdiv((long long)ET * 4, 256), 256, 0, stream>>>(ei, a_s, a_d, emax, den);
  fill_kernel<<<cdiv(NB, 256), 256, 0, stream>>>(B1, 0.0f, (int)NB);
  gat_agg_kernel<<<cdiv((long long)ET * 16, 256), 256, 0, stream>>>(ei, B0, a_s, a_d, emax, den, B1);
  bias_relu_kernel<<<cdiv(NB, 256), 256, 0, stream>>>(B1, bg, 63, (int)NB);

  // --- final: out = relu(h5 @ Wl + bl)   [N,1]
  final_kernel<<<cdiv(N_NODES, 256), 256, 0, stream>>>(B1, Wl, bl, out);
}
